// SparseMultiHeadAttention_88957362635183
// MI455X (gfx1250) — matrix operand order repaired and checked by reading, not compiled, not run on hardware
//
#include <hip/hip_runtime.h>
#include <hip/hip_bf16.h>
#include <math.h>

// ---------------- problem constants (fixed by the reference) ----------------
constexpr int NB   = 2;      // batch
constexpr int S    = 2048;   // sequence
constexpr int D    = 768;    // model dim (= K for all GEMMs)
constexpr int H    = 12;     // heads
constexpr int S2   = S - 2;  // interior rows (G = 1)
constexpr int KIDX = 6;      // 1 global + 2 random + 3 window
constexpr int M_GEMM = NB * S;   // 4096 rows for every GEMM

typedef __attribute__((ext_vector_type(16))) _Float16 v16h;
typedef __attribute__((ext_vector_type(8)))  _Float16 v8h;
typedef __attribute__((ext_vector_type(4)))  _Float16 v4h;
typedef __attribute__((ext_vector_type(8)))  float    v8f;

// ---------------------------------------------------------------------------
// One-time fp32 -> (hi, lo) f16 split: x = hi + lo with |err| ~ x * 2^-22.
// Both planes stored row-major so GEMM fragments are pure b128 loads.
// ---------------------------------------------------------------------------
__global__ __launch_bounds__(256) void pack_hilo(const float* __restrict__ src,
                                                 _Float16* __restrict__ hi,
                                                 _Float16* __restrict__ lo,
                                                 int n4) {
    int i = blockIdx.x * blockDim.x + threadIdx.x;
    if (i >= n4) return;
    float4 x = ((const float4*)src)[i];
    v4h h, l;
    h[0] = (_Float16)x.x; l[0] = (_Float16)(x.x - (float)h[0]);
    h[1] = (_Float16)x.y; l[1] = (_Float16)(x.y - (float)h[1]);
    h[2] = (_Float16)x.z; l[2] = (_Float16)(x.z - (float)h[2]);
    h[3] = (_Float16)x.w; l[3] = (_Float16)(x.w - (float)h[3]);
    ((v4h*)hi)[i] = h;
    ((v4h*)lo)[i] = l;
}

// ---------------------------------------------------------------------------
// Fragment loads from packed row-major f16 (stride D along K):
// A 16x32 (MxK): lane m(<16): elems 0-15 <- K 0..15;
//                lane m+16: K 16..31.
// B 32x16 (KxN): lane n(<16): elems 0-15 <- K 0..15; lane n+16: K 16..31.
// ---------------------------------------------------------------------------
__device__ inline v16h load_afrag(const _Float16* __restrict__ P, int m0, int kk, int lane) {
    const _Float16* p = P + (size_t)(m0 + (lane & 15)) * D + kk + ((lane & 16) ? 16 : 0);
    union { v16h v; v8h h[2]; } u;
    u.h[0] = *(const v8h*)p;        // 16B
    u.h[1] = *(const v8h*)(p + 8);  // 16B
    return u.v;
}

__device__ inline v16h load_bfrag(const _Float16* __restrict__ P, int n0, int kk, int lane) {
    const _Float16* p = P + (size_t)(n0 + (lane & 15)) * D + kk + ((lane & 16) ? 16 : 0);
    return *(const v16h*)p;         // 32B
}

// 3-term split product: C += Ahi*Bhi + Ahi*Blo + Alo*Bhi  (~fp32 accuracy)
__device__ inline v8f wmma3(const v16h& ah, const v16h& al,
                            const v16h& bh, const v16h& bl, v8f c) {
    c = __builtin_amdgcn_wmma_f32_16x16x32_f16(false, ah, false, bh, (short)0, c, false, false);
    c = __builtin_amdgcn_wmma_f32_16x16x32_f16(false, ah, false, bl, (short)0, c, false, false);
    c = __builtin_amdgcn_wmma_f32_16x16x32_f16(false, al, false, bh, (short)0, c, false, false);
    return c;
}

// C/D 16x16 f32 layout: VGPR r, lane l -> M = r + (l<16?0:8), N = l&15
__device__ inline void store_tile(float* __restrict__ C, const v8f& c,
                                  int mt, int nt, int lane, int split_heads) {
    int rbase = mt + ((lane & 16) ? 8 : 0);
    int col   = nt + (lane & 15);
#pragma unroll
    for (int r = 0; r < 8; ++r) {
        int row = rbase + r;
        if (split_heads) {
            // row = b*S + s ; col = h*64 + d ; dest layout [b][h][s][64]
            int b = row >> 11, s = row & (S - 1);
            int h = col >> 6,  d = col & 63;
            C[((((size_t)b * H + h) << 11) + s) * 64 + d] = c[r];
        } else {
            C[(size_t)row * D + col] = c[r];
        }
    }
}

// C(4096x768) = A(4096x768) * W(768x768, NxK)^T, both pre-split into hi/lo f16.
// Block tile 128x128 (8 waves: 4 row x 2 col), wave tile 32x64 (2x4 WMMA tiles).
// Inner loop: 12 x b128-class loads + 24 v_wmma per k-step of 32.
__global__ __launch_bounds__(256) void gemm_nt_wmma(const _Float16* __restrict__ Ah,
                                                    const _Float16* __restrict__ Al,
                                                    const _Float16* __restrict__ Wh,
                                                    const _Float16* __restrict__ Wl,
                                                    float* __restrict__ C,
                                                    int split_heads) {
    const int lane = threadIdx.x & 31;
    const int wid  = threadIdx.x >> 5;
    const int m0   = blockIdx.x * 128 + (wid & 3) * 32;
    const int n0   = blockIdx.y * 128 + (wid >> 2) * 64;

    v8f c[2][4];
#pragma unroll
    for (int i = 0; i < 2; ++i)
#pragma unroll
        for (int j = 0; j < 4; ++j) c[i][j] = (v8f){};

    for (int kk = 0; kk < D; kk += 32) {
        v16h ah[2], al[2], bh[4], bl[4];
#pragma unroll
        for (int i = 0; i < 2; ++i) {
            ah[i] = load_afrag(Ah, m0 + 16 * i, kk, lane);
            al[i] = load_afrag(Al, m0 + 16 * i, kk, lane);
        }
#pragma unroll
        for (int j = 0; j < 4; ++j) {
            bh[j] = load_bfrag(Wh, n0 + 16 * j, kk, lane);
            bl[j] = load_bfrag(Wl, n0 + 16 * j, kk, lane);
        }
#pragma unroll
        for (int i = 0; i < 2; ++i)
#pragma unroll
            for (int j = 0; j < 4; ++j)
                c[i][j] = wmma3(ah[i], al[i], bh[j], bl[j], c[i][j]);
    }
#pragma unroll
    for (int i = 0; i < 2; ++i)
#pragma unroll
        for (int j = 0; j < 4; ++j)
            store_tile(C, c[i][j], m0 + 16 * i, n0 + 16 * j, lane, split_heads);
}

// V_total[b][h][d] = sum_s v[b][h][s][d]   (one block per (b,h))
__global__ __launch_bounds__(256) void vtot_kernel(const float* __restrict__ v,
                                                   float* __restrict__ vtot) {
    __shared__ float red[256];
    const int bh = blockIdx.x, tid = threadIdx.x;
    const int d = tid & 63, chunk = tid >> 6;
    const float* base = v + (size_t)bh * S * 64;
    float part = 0.f;
    for (int t = chunk * 512; t < (chunk + 1) * 512; ++t)
        part += base[(size_t)t * 64 + d];
    red[tid] = part;
    __syncthreads();
    if (tid < 64)
        vtot[(size_t)bh * 64 + d] = red[tid] + red[tid + 64] + red[tid + 128] + red[tid + 192];
}

// Dense attention for the two global rows (pos 0 and S-1). One block per (b,h,row).
__global__ __launch_bounds__(256) void global_rows_kernel(const float* __restrict__ q,
                                                          const float* __restrict__ k,
                                                          const float* __restrict__ v,
                                                          float* __restrict__ heads) {
    __shared__ float sc[S];
    __shared__ float red[256];
    __shared__ float qsh[64];
    const int blk = blockIdx.x;
    const int pos = (blk & 1) ? (S - 1) : 0;
    const int bh  = blk >> 1;
    const int b   = bh / H, h = bh % H;
    const int tid = threadIdx.x;
    const size_t base = ((size_t)b * H + h) * S * 64;

    if (tid < 64) qsh[tid] = q[base + (size_t)pos * 64 + tid];
    __syncthreads();

    float lmax = -3.0e38f;
    for (int t = tid; t < S; t += 256) {
        const float* krow = k + base + (size_t)t * 64;
        float dot = 0.f;
#pragma unroll
        for (int dd = 0; dd < 64; dd += 4) {
            float4 kv = *(const float4*)(krow + dd);
            dot += qsh[dd] * kv.x + qsh[dd + 1] * kv.y + qsh[dd + 2] * kv.z + qsh[dd + 3] * kv.w;
        }
        float s = dot * 0.125f;   // 1/sqrt(64)
        sc[t] = s;
        lmax = fmaxf(lmax, s);
    }
    red[tid] = lmax; __syncthreads();
    for (int st = 128; st > 0; st >>= 1) {
        if (tid < st) red[tid] = fmaxf(red[tid], red[tid + st]);
        __syncthreads();
    }
    float m = red[0];
    __syncthreads();

    float lsum = 0.f;
    for (int t = tid; t < S; t += 256) {
        float e = expf(sc[t] - m);
        sc[t] = e;
        lsum += e;
    }
    red[tid] = lsum; __syncthreads();
    for (int st = 128; st > 0; st >>= 1) {
        if (tid < st) red[tid] += red[tid + st];
        __syncthreads();
    }
    float inv = 1.0f / red[0];
    __syncthreads();

    const int d = tid & 63, chunk = tid >> 6;
    float part = 0.f;
    for (int t = chunk * 512; t < (chunk + 1) * 512; ++t)
        part += sc[t] * v[base + (size_t)t * 64 + d];
    red[tid] = part; __syncthreads();
    if (tid < 64) {
        float o = (red[tid] + red[tid + 64] + red[tid + 128] + red[tid + 192]) * inv;
        heads[((size_t)b * S + pos) * D + h * 64 + d] = o;
    }
}

// Interior rows: 6 sparse keys + softmax over a mostly-ZERO 2048-wide row.
// out = p0 * Vtot + sum_j (p_j - p0) * v[idx_j]   (one wave per (b,h,pos))
__global__ __launch_bounds__(256) void sparse_rows_kernel(const float* __restrict__ q,
                                                          const float* __restrict__ k,
                                                          const float* __restrict__ v,
                                                          const float* __restrict__ vtot,
                                                          const int* __restrict__ idx,
                                                          float* __restrict__ heads) {
    const int gw   = (blockIdx.x * blockDim.x + threadIdx.x) >> 5;
    const int lane = threadIdx.x & 31;
    const int total = NB * H * S2;
    if (gw >= total) return;
    const int b   = gw / (H * S2);
    const int rem = gw % (H * S2);
    const int h   = rem / S2;
    const int i   = rem % S2;
    const int pos = i + 1;

    const size_t base = ((size_t)b * H + h) * S * 64;
    const float2 q2 = *(const float2*)(q + base + (size_t)pos * 64 + lane * 2);

    int   kid[KIDX];
    float sc[KIDX];
    bool  valid[KIDX];
#pragma unroll
    for (int j = 0; j < KIDX; ++j) {
        kid[j] = idx[i * KIDX + j];
        const float2 k2 = *(const float2*)(k + base + (size_t)kid[j] * 64 + lane * 2);
        float p = q2.x * k2.x + q2.y * k2.y;
#pragma unroll
        for (int off = 16; off > 0; off >>= 1) p += __shfl_xor(p, off, 32);
        sc[j] = p * 0.125f;
    }
    int ndist = 0;
#pragma unroll
    for (int j = 0; j < KIDX; ++j) {
        valid[j] = true;
#pragma unroll
        for (int jj = 0; jj < KIDX; ++jj)
            if (jj < j && kid[jj] == kid[j]) valid[j] = false;
        if (valid[j]) ndist++;
    }
    float m = 0.f;  // the (S - ndist) zero entries participate in the max
#pragma unroll
    for (int j = 0; j < KIDX; ++j)
        if (valid[j]) m = fmaxf(m, sc[j]);

    float ez = expf(-m);
    float denom = (float)(S - ndist) * ez;
    float e[KIDX];
#pragma unroll
    for (int j = 0; j < KIDX; ++j) {
        e[j] = valid[j] ? expf(sc[j] - m) : 0.f;
        denom += e[j];
    }
    const float p0 = ez / denom;

    const float* vt = vtot + ((size_t)b * H + h) * 64;
    float2 acc;
    acc.x = p0 * vt[lane * 2];
    acc.y = p0 * vt[lane * 2 + 1];
#pragma unroll
    for (int j = 0; j < KIDX; ++j) {
        if (!valid[j]) continue;
        float w = e[j] / denom - p0;
        const float2 v2 = *(const float2*)(v + base + (size_t)kid[j] * 64 + lane * 2);
        acc.x += w * v2.x;
        acc.y += w * v2.y;
    }
    float* o = heads + ((size_t)b * S + pos) * D + h * 64 + lane * 2;
    o[0] = acc.x;
    o[1] = acc.y;
}

// ---------------------------------------------------------------------------
extern "C" void kernel_launch(void* const* d_in, const int* in_sizes, int n_in,
                              void* d_out, int out_size, void* d_ws, size_t ws_size,
                              hipStream_t stream) {
    const float* Q  = (const float*)d_in[0];
    const float* K  = (const float*)d_in[1];
    const float* V  = (const float*)d_in[2];
    const float* Wq = (const float*)d_in[3];
    const float* Wk = (const float*)d_in[4];
    const float* Wv = (const float*)d_in[5];
    const float* Wo = (const float*)d_in[6];
    const int*   idx = (const int*)d_in[7];
    float* out = (float*)d_out;

    const size_t PER = (size_t)NB * H * S * 64;   // 3,145,728 floats
    float* qws   = (float*)d_ws;
    float* kws   = qws + PER;
    float* vws   = kws + PER;
    float* heads = vws + PER;                     // [b][s][h*64+d]
    float* vtot  = heads + (size_t)NB * S * D;    // [b][h][64]
    // f16 staging planes (reused across the serialized GEMMs)
    _Float16* Ahi = (_Float16*)(vtot + NB * H * 64);
    _Float16* Alo = Ahi + (size_t)M_GEMM * D;
    _Float16* Whi = Alo + (size_t)M_GEMM * D;
    _Float16* Wlo = Whi + (size_t)D * D;

    const int n4A = M_GEMM * D / 4;               // 786,432 float4s
    const int n4W = D * D / 4;                    // 147,456 float4s
    dim3 gg(M_GEMM / 128, D / 128);

    // q = Q @ Wq^T (head-split layout)
    pack_hilo<<<(n4A + 255) / 256, 256, 0, stream>>>(Q, Ahi, Alo, n4A);
    pack_hilo<<<(n4W + 255) / 256, 256, 0, stream>>>(Wq, Whi, Wlo, n4W);
    gemm_nt_wmma<<<gg, 256, 0, stream>>>(Ahi, Alo, Whi, Wlo, qws, 1);
    // k = K @ Wk^T
    pack_hilo<<<(n4A + 255) / 256, 256, 0, stream>>>(K, Ahi, Alo, n4A);
    pack_hilo<<<(n4W + 255) / 256, 256, 0, stream>>>(Wk, Whi, Wlo, n4W);
    gemm_nt_wmma<<<gg, 256, 0, stream>>>(Ahi, Alo, Whi, Wlo, kws, 1);
    // v = V @ Wv^T
    pack_hilo<<<(n4A + 255) / 256, 256, 0, stream>>>(V, Ahi, Alo, n4A);
    pack_hilo<<<(n4W + 255) / 256, 256, 0, stream>>>(Wv, Whi, Wlo, n4W);
    gemm_nt_wmma<<<gg, 256, 0, stream>>>(Ahi, Alo, Whi, Wlo, vws, 1);

    vtot_kernel<<<NB * H, 256, 0, stream>>>(vws, vtot);
    global_rows_kernel<<<NB * H * 2, 256, 0, stream>>>(qws, kws, vws, heads);

    const int rows = NB * H * S2;                 // 49104 waves, 8 per block
    sparse_rows_kernel<<<(rows + 7) / 8, 256, 0, stream>>>(qws, kws, vws, vtot, idx, heads);

    // out = heads @ Wo^T (plain layout)
    pack_hilo<<<(n4A + 255) / 256, 256, 0, stream>>>(heads, Ahi, Alo, n4A);
    pack_hilo<<<(n4W + 255) / 256, 256, 0, stream>>>(Wo, Whi, Wlo, n4W);
    gemm_nt_wmma<<<gg, 256, 0, stream>>>(Ahi, Alo, Whi, Wlo, out, 0);
}
